// SequentialMultiHeadAttention_50062138802613
// MI455X (gfx1250) — matrix operand order repaired and checked by reading, not compiled, not run on hardware
//
#include <hip/hip_runtime.h>

#define TSEQ 2048
#define EMB  1024
#define NH   16
#define HD   64
#define NB   4

typedef __attribute__((ext_vector_type(16))) __bf16 v16bf;
typedef __attribute__((ext_vector_type(8)))  float  v8f;

__device__ __forceinline__ v8f wmma_bf16(v16bf a, v16bf b, v8f c) {
  // (neg_a, A, neg_b, B, c_mod, C, reuse_a, reuse_b)
  return __builtin_amdgcn_wmma_f32_16x16x32_bf16(false, a, false, b, (short)0, c, false, false);
}

// ---------------------------------------------------------------------------
// Kernel 1: convert+transpose per-head weights [H,E,HD] f32 -> [H,HD,E] bf16
// so WMMA B-fragments (K contiguous) are two 16B loads per lane.
// ---------------------------------------------------------------------------
__global__ void prep_w_kernel(const float* __restrict__ W, __bf16* __restrict__ WT) {
  int idx = blockIdx.x * blockDim.x + threadIdx.x;  // over NH*HD*EMB
  if (idx >= NH * HD * EMB) return;
  int e = idx & (EMB - 1);
  int d = (idx >> 10) & (HD - 1);
  int h = idx >> 16;
  WT[idx] = (__bf16)W[((size_t)h * EMB + e) * HD + d];
}

// ---------------------------------------------------------------------------
// Kernel 2: QKV projection. grid(T/16, H, B), block 128 (4 waves).
// Wave w computes the 16-wide d-tile w of q,k,v for one 16-row t tile.
// q,k stored [B,H,T,HD] bf16; v stored transposed [B,H,HD,T] bf16.
// ---------------------------------------------------------------------------
__global__ void __launch_bounds__(128) qkv_kernel(
    const float* __restrict__ x,
    const __bf16* __restrict__ WqT, const __bf16* __restrict__ WkT,
    const __bf16* __restrict__ WvT,
    __bf16* __restrict__ q, __bf16* __restrict__ k, __bf16* __restrict__ vT) {
  const int wave = threadIdx.x >> 5;
  const int lane = threadIdx.x & 31;
  const int tTile = blockIdx.x;
  const int h = blockIdx.y;
  const int b = blockIdx.z;
  const int row = lane & 15;             // A: M row ; B/D: N col
  const int kbA = (lane >> 4) * 16;      // A-frag K base (contiguous halves)
  const int kbB = (lane >> 4) * 16;      // B-frag K base (contiguous halves)
  const int dcol = wave * 16 + row;

  const float*  xrow = x + ((size_t)b * TSEQ + tTile * 16 + row) * EMB;
  const __bf16* bq = WqT + ((size_t)h * HD + dcol) * EMB;
  const __bf16* bk = WkT + ((size_t)h * HD + dcol) * EMB;
  const __bf16* bv = WvT + ((size_t)h * HD + dcol) * EMB;

  v8f accq = {}, acck = {}, accv = {};
  for (int k0 = 0; k0 < EMB; k0 += 32) {
    v16bf A, Bq, Bk, Bv;
#pragma unroll
    for (int e = 0; e < 8; e++) {
      A[e]     = (__bf16)xrow[k0 + kbA + e];
      A[e + 8] = (__bf16)xrow[k0 + 8 + kbA + e];
    }
#pragma unroll
    for (int e = 0; e < 16; e++) {
      Bq[e] = bq[k0 + kbB + e];
      Bk[e] = bk[k0 + kbB + e];
      Bv[e] = bv[k0 + kbB + e];
    }
    accq = wmma_bf16(A, Bq, accq);
    acck = wmma_bf16(A, Bk, acck);
    accv = wmma_bf16(A, Bv, accv);
  }
  const int Mb = (lane >> 4) * 8;
  const size_t bh = (size_t)b * NH + h;
#pragma unroll
  for (int r = 0; r < 8; r++) {
    const int t = tTile * 16 + Mb + r;
    const size_t qi = (bh * TSEQ + t) * HD + dcol;   // N = dcol here (row==n)
    q[qi] = (__bf16)accq[r];
    k[qi] = (__bf16)acck[r];
    vT[(bh * HD + dcol) * TSEQ + t] = (__bf16)accv[r];
  }
}

// ---------------------------------------------------------------------------
// Kernel 3: flash attention. grid(T/64, H, B), block 128; each wave owns a
// 16-query tile, streams 32-wide key blocks with online softmax.
// ---------------------------------------------------------------------------
__global__ void __launch_bounds__(128) attn_kernel(
    const __bf16* __restrict__ q, const __bf16* __restrict__ kmat,
    const __bf16* __restrict__ vT, __bf16* __restrict__ attnOut) {
  __shared__ __bf16 pbuf[4][16][40];  // per-wave P transpose staging (D->A)
  const int wave = threadIdx.x >> 5;
  const int lane = threadIdx.x & 31;
  const int tTile = blockIdx.x * 4 + wave;
  const int h = blockIdx.y;
  const int b = blockIdx.z;
  const int t_base = tTile * 16;
  const int row = lane & 15;
  const int kbA = (lane >> 4) * 16;
  const int kbB = (lane >> 4) * 16;
  const int n = lane & 15;
  const int Mb = (lane >> 4) * 8;
  const size_t bh = (size_t)b * NH + h;

  // Q tile as two A fragments (K = d = 0..31, 32..63)
  const __bf16* qrow = q + (bh * TSEQ + t_base + row) * HD;
  v16bf A0, A1;
#pragma unroll
  for (int e = 0; e < 8; e++) {
    A0[e]     = qrow[kbA + e];
    A0[e + 8] = qrow[8 + kbA + e];
    A1[e]     = qrow[32 + kbA + e];
    A1[e + 8] = qrow[40 + kbA + e];
  }

  const __bf16* kbase = kmat + bh * TSEQ * HD;
  const __bf16* vbase = vT + bh * HD * TSEQ;

  v8f acc[4] = {v8f{}, v8f{}, v8f{}, v8f{}};
  float mrow[8], lrow[8];
#pragma unroll
  for (int r = 0; r < 8; r++) { mrow[r] = -__builtin_inff(); lrow[r] = 0.f; }

  const int jmax = (t_base + 15) >> 5;   // only block jmax is partially masked
  for (int j = 0; j <= jmax; j++) {
    const int s0 = j * 32;
    // K^T B fragments: B[kdim=d][ncol=s]; k rows contiguous in d.
    const __bf16* kp0 = kbase + (size_t)(s0 + n) * HD;
    const __bf16* kp1 = kbase + (size_t)(s0 + 16 + n) * HD;
    v16bf B0a, B0b, B1a, B1b;
#pragma unroll
    for (int e = 0; e < 16; e++) {
      B0a[e] = kp0[kbB + e];
      B0b[e] = kp0[32 + kbB + e];
      B1a[e] = kp1[kbB + e];
      B1b[e] = kp1[32 + kbB + e];
    }
    v8f s0t = {}, s1t = {};
    s0t = wmma_bf16(A0, B0a, s0t);
    s0t = wmma_bf16(A1, B0b, s0t);
    s1t = wmma_bf16(A0, B1a, s1t);
    s1t = wmma_bf16(A1, B1b, s1t);

    const bool maskblk = (j == jmax);
    float p0[8], p1[8];
#pragma unroll
    for (int r = 0; r < 8; r++) {
      float v0 = s0t[r] * 0.125f;        // HD^-0.5
      float v1 = s1t[r] * 0.125f;
      if (maskblk) {                      // lane-local select; EXEC stays full
        const int trow = t_base + Mb + r;
        if (s0 + n > trow)      v0 = -__builtin_inff();
        if (s0 + 16 + n > trow) v1 = -__builtin_inff();
      }
      float mx = fmaxf(v0, v1);
#pragma unroll
      for (int off = 1; off < 16; off <<= 1)
        mx = fmaxf(mx, __shfl_xor(mx, off, 16));
      const float mnew = fmaxf(mrow[r], mx);
      const float alpha = __expf(mrow[r] - mnew);
      mrow[r] = mnew;
      v0 = __expf(v0 - mnew);
      v1 = __expf(v1 - mnew);
      float sum = v0 + v1;
#pragma unroll
      for (int off = 1; off < 16; off <<= 1)
        sum += __shfl_xor(sum, off, 16);
      lrow[r] = lrow[r] * alpha + sum;
#pragma unroll
      for (int dt = 0; dt < 4; dt++) acc[dt][r] *= alpha;
      p0[r] = v0;
      p1[r] = v1;
    }

    // D-layout -> A-layout for P via per-wave LDS tile (same-wave RAW,
    // compiler inserts s_wait_dscnt).
#pragma unroll
    for (int r = 0; r < 8; r++) {
      pbuf[wave][Mb + r][n]      = (__bf16)p0[r];
      pbuf[wave][Mb + r][16 + n] = (__bf16)p1[r];
    }
    v16bf PA;
#pragma unroll
    for (int e = 0; e < 8; e++) {
      PA[e]     = pbuf[wave][row][kbA + e];
      PA[e + 8] = pbuf[wave][row][8 + kbA + e];
    }
    // O += P @ V : vT rows contiguous in s (K dim).
#pragma unroll
    for (int dt = 0; dt < 4; dt++) {
      const __bf16* vp = vbase + (size_t)(dt * 16 + n) * TSEQ + s0;
      v16bf BV;
#pragma unroll
      for (int e = 0; e < 16; e++) BV[e] = vp[kbB + e];
      acc[dt] = wmma_bf16(PA, BV, acc[dt]);
    }
  }

  // normalize and store concat-head layout [B,T,E] bf16
#pragma unroll
  for (int r = 0; r < 8; r++) {
    const float inv = 1.0f / lrow[r];
    const int t = t_base + Mb + r;
#pragma unroll
    for (int dt = 0; dt < 4; dt++) {
      attnOut[((size_t)b * TSEQ + t) * EMB + h * HD + dt * 16 + n] =
          (__bf16)(acc[dt][r] * inv);
    }
  }
}

// ---------------------------------------------------------------------------
// Kernel 4: output projection y = attn @ Wo^T + bo.
// grid(B*T/16, E/64), block 128; wave w -> N-tile blockIdx.y*4+w.
// B[k][n] = Wo[n][k] -> contiguous in k, cvt f32->bf16 on the fly.
// ---------------------------------------------------------------------------
__global__ void __launch_bounds__(128) proj_kernel(
    const __bf16* __restrict__ attn, const float* __restrict__ Wo,
    const float* __restrict__ bo, float* __restrict__ y) {
  const int wave = threadIdx.x >> 5;
  const int lane = threadIdx.x & 31;
  const int mTile = blockIdx.x;
  const int nTile = blockIdx.y * 4 + wave;
  const int row = lane & 15;
  const int kbA = (lane >> 4) * 16;
  const int kbB = (lane >> 4) * 16;
  const int n = lane & 15;

  const __bf16* arow = attn + ((size_t)mTile * 16 + row) * EMB;
  const float*  wrow = Wo + (size_t)(nTile * 16 + n) * EMB;
  v8f acc = {};
  for (int k0 = 0; k0 < EMB; k0 += 32) {
    v16bf A, Bm;
#pragma unroll
    for (int e = 0; e < 8; e++) {
      A[e]     = arow[k0 + kbA + e];
      A[e + 8] = arow[k0 + 8 + kbA + e];
    }
#pragma unroll
    for (int e = 0; e < 16; e++) Bm[e] = (__bf16)wrow[k0 + kbB + e];
    acc = wmma_bf16(A, Bm, acc);
  }
  const int Mb = (lane >> 4) * 8;
  const float bias = bo[nTile * 16 + n];
#pragma unroll
  for (int r = 0; r < 8; r++) {
    y[((size_t)mTile * 16 + Mb + r) * EMB + nTile * 16 + n] = acc[r] + bias;
  }
}

// ---------------------------------------------------------------------------
// Workspace layout (bytes), total 70 MB:
//   [0,2M)   WqT bf16   [2M,4M)  WkT   [4M,6M)  WvT
//   [6M,22M) q bf16     [22M,38M) k    [38M,54M) vT   [54M,70M) attn bf16
// ---------------------------------------------------------------------------
extern "C" void kernel_launch(void* const* d_in, const int* in_sizes, int n_in,
                              void* d_out, int out_size, void* d_ws, size_t ws_size,
                              hipStream_t stream) {
  const float* x  = (const float*)d_in[0];
  const float* Wq = (const float*)d_in[1];
  const float* Wk = (const float*)d_in[2];
  const float* Wv = (const float*)d_in[3];
  const float* Wo = (const float*)d_in[4];
  const float* bo = (const float*)d_in[5];
  float* y = (float*)d_out;

  char* ws = (char*)d_ws;
  const size_t szWT = (size_t)NH * HD * EMB * 2;        // 2 MB
  const size_t szQ  = (size_t)NB * NH * TSEQ * HD * 2;  // 16 MB
  __bf16* WqT = (__bf16*)(ws);
  __bf16* WkT = (__bf16*)(ws + szWT);
  __bf16* WvT = (__bf16*)(ws + 2 * szWT);
  __bf16* qb  = (__bf16*)(ws + 3 * szWT);
  __bf16* kb  = (__bf16*)(ws + 3 * szWT + szQ);
  __bf16* vTb = (__bf16*)(ws + 3 * szWT + 2 * szQ);
  __bf16* ab  = (__bf16*)(ws + 3 * szWT + 3 * szQ);

  const int nprep = NH * HD * EMB;
  prep_w_kernel<<<(nprep + 255) / 256, 256, 0, stream>>>(Wq, WqT);
  prep_w_kernel<<<(nprep + 255) / 256, 256, 0, stream>>>(Wk, WkT);
  prep_w_kernel<<<(nprep + 255) / 256, 256, 0, stream>>>(Wv, WvT);

  qkv_kernel<<<dim3(TSEQ / 16, NH, NB), 128, 0, stream>>>(x, WqT, WkT, WvT, qb, kb, vTb);
  attn_kernel<<<dim3(TSEQ / 64, NH, NB), 128, 0, stream>>>(qb, kb, vTb, ab);
  proj_kernel<<<dim3(NB * TSEQ / 16, EMB / 64), 128, 0, stream>>>(ab, Wo, bo, y);
}
